// MomentPredictor_5085241278776
// MI455X (gfx1250) — matrix operand order repaired and checked by reading, not compiled, not run on hardware
//
#include <hip/hip_runtime.h>

namespace {

constexpr int B_ = 512, T_ = 16, DIN_ = 7, D_ = 512, L_ = 2, H_ = 8, FF_ = 2048, HD_ = 64;

typedef __attribute__((ext_vector_type(16))) __bf16          v16bf;
typedef __attribute__((ext_vector_type(8)))  float           v8f;
typedef __attribute__((ext_vector_type(8)))  unsigned short  v8u16;
typedef __attribute__((ext_vector_type(16))) unsigned short  v16u16;

__device__ __forceinline__ unsigned short f2bf(float f) {
  unsigned u = __builtin_bit_cast(unsigned, f);
  u += 0x7FFFu + ((u >> 16) & 1u);           // round-to-nearest-even
  return (unsigned short)(u >> 16);
}
__device__ __forceinline__ float bf2f(unsigned short h) {
  unsigned u = ((unsigned)h) << 16;
  return __builtin_bit_cast(float, u);
}

// A fragment (16x32 bf16, row-major source [M][K]):
// lane holds row m0+(lane&15); element i has K = k0 + i + 16*(lane>>4)
// -> two aligned 16B chunks at +16*hi and +16*hi+8.
__device__ __forceinline__ v16bf load_a_frag(const unsigned short* p) {
  v8u16 c0 = *(const v8u16*)p;
  v8u16 c1 = *(const v8u16*)(p + 8);
  v16u16 m = __builtin_shufflevector(c0, c1, 0,1,2,3,4,5,6,7,8,9,10,11,12,13,14,15);
  return __builtin_bit_cast(v16bf, m);
}
// B fragment (32x16 bf16) from pre-transposed Wt [N][K]:
// lane holds col n0+(lane&15), contiguous K run of 16 starting at k0+16*(lane>>4).
__device__ __forceinline__ v16bf load_b_frag(const unsigned short* p) {
  v8u16 c0 = *(const v8u16*)p;
  v8u16 c1 = *(const v8u16*)(p + 8);
  v16u16 m = __builtin_shufflevector(c0, c1, 0,1,2,3,4,5,6,7,8,9,10,11,12,13,14,15);
  return __builtin_bit_cast(v16bf, m);
}

// ---------------- bf16 GEMM: C[M,N] = A[M,K] @ Wt[N,K]^T + bias ----------------
// block = 128 threads = 4 waves (2x2), each wave a 32x32 tile via 4 WMMA accumulators.
template <bool RELU>
__global__ __launch_bounds__(128)
void gemm_kernel(const unsigned short* __restrict__ A,
                 const unsigned short* __restrict__ Wt,
                 const float* __restrict__ bias,
                 float* Cf, unsigned short* Cb,
                 int M, int N, int K) {
  const int lane = threadIdx.x & 31;
  const int wave = threadIdx.x >> 5;
  const int tileM = blockIdx.x * 64 + (wave & 1) * 32;
  const int tileN = blockIdx.y * 64 + (wave >> 1) * 32;

  const v8f vzero = {0.f, 0.f, 0.f, 0.f, 0.f, 0.f, 0.f, 0.f};
  v8f acc[2][2];
  acc[0][0] = vzero; acc[0][1] = vzero; acc[1][0] = vzero; acc[1][1] = vzero;

  const int l15 = lane & 15;
  const int hi  = lane >> 4;
  const unsigned short* a0 = A  + (size_t)(tileM      + l15) * K + hi * 16;
  const unsigned short* a1 = A  + (size_t)(tileM + 16 + l15) * K + hi * 16;
  const unsigned short* b0 = Wt + (size_t)(tileN      + l15) * K + hi * 16;
  const unsigned short* b1 = Wt + (size_t)(tileN + 16 + l15) * K + hi * 16;

  for (int k = 0; k < K; k += 32) {
    v16bf af0 = load_a_frag(a0 + k);
    v16bf af1 = load_a_frag(a1 + k);
    v16bf bf0 = load_b_frag(b0 + k);
    v16bf bf1 = load_b_frag(b1 + k);
    acc[0][0] = __builtin_amdgcn_wmma_f32_16x16x32_bf16(false, af0, false, bf0, (short)0, acc[0][0], false, false);
    acc[0][1] = __builtin_amdgcn_wmma_f32_16x16x32_bf16(false, af0, false, bf1, (short)0, acc[0][1], false, false);
    acc[1][0] = __builtin_amdgcn_wmma_f32_16x16x32_bf16(false, af1, false, bf0, (short)0, acc[1][0], false, false);
    acc[1][1] = __builtin_amdgcn_wmma_f32_16x16x32_bf16(false, af1, false, bf1, (short)0, acc[1][1], false, false);
  }

  // D layout: lane -> col tileN+(lane&15); VGPR r -> row 8*(lane>>4)+r
  const int col0 = tileN + l15;
  const int row0 = tileM + hi * 8;
#pragma unroll
  for (int j = 0; j < 2; ++j) {
    const int col = col0 + j * 16;
    const float bv = bias ? bias[col] : 0.f;
#pragma unroll
    for (int i = 0; i < 2; ++i) {
      const int rb = row0 + i * 16;
#pragma unroll
      for (int r = 0; r < 8; ++r) {
        float v = acc[i][j][r] + bv;
        if (RELU) v = v > 0.f ? v : 0.f;
        if (Cf) Cf[(size_t)(rb + r) * N + col] = v;
        if (Cb) Cb[(size_t)(rb + r) * N + col] = f2bf(v);
      }
    }
  }
}

// -------- weight convert+transpose: src f32 [K][N] -> dst bf16 [N][K] --------
__global__ void convert_wt_kernel(const float* __restrict__ src,
                                  unsigned short* __restrict__ dst,
                                  int K, int N) {
  int idx = blockIdx.x * 256 + threadIdx.x;
  if (idx >= K * N) return;
  int n = idx / K, k = idx % K;
  dst[idx] = f2bf(src[(size_t)k * N + n]);
}

// -------- embed: h = x @ ip_w + ip_b + pe[t]; feedback features from d_out --------
__global__ __launch_bounds__(128)
void embed_kernel(const float* __restrict__ x_in, const float* __restrict__ ip_w,
                  const float* __restrict__ ip_b, const float* __restrict__ y_prev,
                  int t, float* __restrict__ h_f32, unsigned short* __restrict__ h_bf) {
  const int b = blockIdx.x;
  float x[DIN_];
  const float* xr = x_in + ((size_t)b * T_ + t) * DIN_;
#pragma unroll
  for (int i = 0; i < DIN_; ++i) x[i] = xr[i];
  if (t > 0) {
#pragma unroll
    for (int c = 0; c < 3; ++c) x[DIN_ - 3 + c] = y_prev[((size_t)b * T_ + (t - 1)) * 3 + c];
  }
  for (int d = threadIdx.x; d < D_; d += 128) {
    float acc = ip_b[d];
#pragma unroll
    for (int i = 0; i < DIN_; ++i) acc += x[i] * ip_w[i * D_ + d];
    // pe: div = exp((d&~1) * (-ln(1e4)/512)); even->sin, odd->cos
    float ang = (float)t * __expf((float)(d & ~1) * -0.0179889453f);
    acc += (d & 1) ? __cosf(ang) : __sinf(ang);
    h_f32[(size_t)b * D_ + d] = acc;
    h_bf[(size_t)b * D_ + d] = f2bf(acc);
  }
}

// -------- scatter K/V from qkv_f32 into bf16 caches [B][H][T][HD] --------
__global__ void scatter_kv_kernel(const float* __restrict__ qkv,
                                  unsigned short* __restrict__ kc,
                                  unsigned short* __restrict__ vc, int t) {
  int idx = blockIdx.x * 256 + threadIdx.x;
  if (idx >= B_ * D_) return;
  int b = idx / D_, c = idx % D_;
  int h = c / HD_, d = c % HD_;
  size_t o = (((size_t)(b * H_ + h)) * T_ + t) * HD_ + d;
  kc[o] = f2bf(qkv[(size_t)b * (3 * D_) +     D_ + c]);
  vc[o] = f2bf(qkv[(size_t)b * (3 * D_) + 2 * D_ + c]);
}

// -------- attention: one wave32 per (b,h); <=16 keys, HD=64 split over lanes --------
__global__ __launch_bounds__(256)
void attn_kernel(const float* __restrict__ qkv, const unsigned short* __restrict__ kc,
                 const unsigned short* __restrict__ vc, unsigned short* __restrict__ ao, int t) {
  int gid = blockIdx.x * 256 + threadIdx.x;
  int wid = gid >> 5;                 // (b*H + h)
  int lane = gid & 31;
  int b = wid / H_, h = wid % H_;
  float q0 = qkv[(size_t)b * (3 * D_) + h * HD_ + lane];
  float q1 = qkv[(size_t)b * (3 * D_) + h * HD_ + 32 + lane];
  const unsigned short* kb = kc + ((size_t)(b * H_ + h)) * T_ * HD_;
  const unsigned short* vb = vc + ((size_t)(b * H_ + h)) * T_ * HD_;
  float s[T_];
#pragma unroll
  for (int j = 0; j < T_; ++j) {
    float p = 0.f;
    if (j <= t) p = q0 * bf2f(kb[j * HD_ + lane]) + q1 * bf2f(kb[j * HD_ + 32 + lane]);
#pragma unroll
    for (int off = 16; off > 0; off >>= 1) p += __shfl_xor(p, off, 32);
    s[j] = (j <= t) ? p * 0.125f : -1e30f;   // 1/sqrt(64)
  }
  float m = s[0];
#pragma unroll
  for (int j = 1; j < T_; ++j) m = fmaxf(m, s[j]);
  float sum = 0.f;
#pragma unroll
  for (int j = 0; j < T_; ++j) { s[j] = __expf(s[j] - m); sum += s[j]; }
  float inv = 1.f / sum;
  float o0 = 0.f, o1 = 0.f;
#pragma unroll
  for (int j = 0; j < T_; ++j) {
    if (j <= t) {
      float w = s[j] * inv;
      o0 += w * bf2f(vb[j * HD_ + lane]);
      o1 += w * bf2f(vb[j * HD_ + 32 + lane]);
    }
  }
  ao[(size_t)b * D_ + h * HD_ + lane]      = f2bf(o0);
  ao[(size_t)b * D_ + h * HD_ + 32 + lane] = f2bf(o1);
}

// -------- residual add + LayerNorm, writes f32 + bf16 (in-place safe per-row) --------
__global__ __launch_bounds__(128)
void add_ln_kernel(const float* hin, const float* __restrict__ delta,
                   const float* __restrict__ gamma, const float* __restrict__ beta,
                   float* hout, unsigned short* __restrict__ hbf) {
  const int b = blockIdx.x;
  __shared__ float red[128];
  float x[4];
  float lsum = 0.f;
#pragma unroll
  for (int i = 0; i < 4; ++i) {
    int d = threadIdx.x + i * 128;
    x[i] = hin[(size_t)b * D_ + d] + delta[(size_t)b * D_ + d];
    lsum += x[i];
  }
  red[threadIdx.x] = lsum;
  __syncthreads();
  for (int s2 = 64; s2 > 0; s2 >>= 1) {
    if (threadIdx.x < s2) red[threadIdx.x] += red[threadIdx.x + s2];
    __syncthreads();
  }
  const float mean = red[0] * (1.f / (float)D_);
  __syncthreads();
  float lvar = 0.f;
#pragma unroll
  for (int i = 0; i < 4; ++i) { float dd = x[i] - mean; lvar += dd * dd; }
  red[threadIdx.x] = lvar;
  __syncthreads();
  for (int s2 = 64; s2 > 0; s2 >>= 1) {
    if (threadIdx.x < s2) red[threadIdx.x] += red[threadIdx.x + s2];
    __syncthreads();
  }
  const float rstd = rsqrtf(red[0] * (1.f / (float)D_) + 1e-5f);
#pragma unroll
  for (int i = 0; i < 4; ++i) {
    int d = threadIdx.x + i * 128;
    float y = (x[i] - mean) * rstd * gamma[d] + beta[d];
    hout[(size_t)b * D_ + d] = y;
    hbf[(size_t)b * D_ + d] = f2bf(y);
  }
}

// -------- head: y[b,c] = h[b,:] @ head_w[:,c] + head_b[c] --------
__global__ void head_kernel(const float* __restrict__ h, const float* __restrict__ hw,
                            const float* __restrict__ hb, float* __restrict__ out, int t) {
  int idx = blockIdx.x * 256 + threadIdx.x;
  if (idx >= B_ * 3) return;
  int b = idx / 3, c = idx % 3;
  float s = hb[c];
  for (int d = 0; d < D_; ++d) s += h[(size_t)b * D_ + d] * hw[d * 3 + c];
  out[((size_t)b * T_ + t) * 3 + c] = s;
}

} // namespace

extern "C" void kernel_launch(void* const* d_in, const int* in_sizes, int n_in,
                              void* d_out, int out_size, void* d_ws, size_t ws_size,
                              hipStream_t stream) {
  (void)in_sizes; (void)n_in; (void)out_size; (void)ws_size;
  const float* x_in   = (const float*)d_in[0];
  const float* ip_w   = (const float*)d_in[1];
  const float* ip_b   = (const float*)d_in[2];
  const float* qkv_w  = (const float*)d_in[3];
  const float* qkv_b  = (const float*)d_in[4];
  const float* out_w  = (const float*)d_in[5];
  const float* out_b  = (const float*)d_in[6];
  const float* ln1_s  = (const float*)d_in[7];
  const float* ln1_b  = (const float*)d_in[8];
  const float* ff_w1  = (const float*)d_in[9];
  const float* ff_b1  = (const float*)d_in[10];
  const float* ff_w2  = (const float*)d_in[11];
  const float* ff_b2  = (const float*)d_in[12];
  const float* ln2_s  = (const float*)d_in[13];
  const float* ln2_b  = (const float*)d_in[14];
  const float* head_w = (const float*)d_in[15];
  const float* head_b = (const float*)d_in[16];
  float* out = (float*)d_out;

  // ---- workspace carve-up (aligned to 256B) ----
  char* ws = (char*)d_ws;
  auto alloc = [&](size_t bytes) -> char* {
    char* p = ws;
    ws += (bytes + 255) & ~(size_t)255;
    return p;
  };
  unsigned short* wt_qkv = (unsigned short*)alloc((size_t)L_ * 3 * D_ * D_ * 2);
  unsigned short* wt_out = (unsigned short*)alloc((size_t)L_ * D_ * D_ * 2);
  unsigned short* wt_ff1 = (unsigned short*)alloc((size_t)L_ * FF_ * D_ * 2);
  unsigned short* wt_ff2 = (unsigned short*)alloc((size_t)L_ * D_ * FF_ * 2);
  unsigned short* kc     = (unsigned short*)alloc((size_t)L_ * B_ * H_ * T_ * HD_ * 2);
  unsigned short* vc     = (unsigned short*)alloc((size_t)L_ * B_ * H_ * T_ * HD_ * 2);
  float*          h_f32  = (float*)alloc((size_t)B_ * D_ * 4);
  unsigned short* h_bf   = (unsigned short*)alloc((size_t)B_ * D_ * 2);
  float*          qkv_f  = (float*)alloc((size_t)B_ * 3 * D_ * 4);
  unsigned short* ao_bf  = (unsigned short*)alloc((size_t)B_ * D_ * 2);
  unsigned short* ffm_bf = (unsigned short*)alloc((size_t)B_ * FF_ * 2);
  float*          dlt    = (float*)alloc((size_t)B_ * D_ * 4);
  const size_t cacheL = (size_t)B_ * H_ * T_ * HD_;

  // ---- one-time weight convert+transpose to bf16 [N][K] (L2-resident thereafter) ----
  for (int l = 0; l < L_; ++l) {
    convert_wt_kernel<<<(D_ * 3 * D_ + 255) / 256, 256, 0, stream>>>(
        qkv_w + (size_t)l * D_ * 3 * D_, wt_qkv + (size_t)l * 3 * D_ * D_, D_, 3 * D_);
    convert_wt_kernel<<<(D_ * D_ + 255) / 256, 256, 0, stream>>>(
        out_w + (size_t)l * D_ * D_, wt_out + (size_t)l * D_ * D_, D_, D_);
    convert_wt_kernel<<<(D_ * FF_ + 255) / 256, 256, 0, stream>>>(
        ff_w1 + (size_t)l * D_ * FF_, wt_ff1 + (size_t)l * FF_ * D_, D_, FF_);
    convert_wt_kernel<<<(FF_ * D_ + 255) / 256, 256, 0, stream>>>(
        ff_w2 + (size_t)l * FF_ * D_, wt_ff2 + (size_t)l * D_ * FF_, FF_, D_);
  }

  // ---- autoregressive rollout with exact KV caching ----
  for (int t = 0; t < T_; ++t) {
    embed_kernel<<<B_, 128, 0, stream>>>(x_in, ip_w, ip_b, out, t, h_f32, h_bf);
    for (int l = 0; l < L_; ++l) {
      // qkv = h @ qkv_w + qkv_b   : [512,512]x[512,1536]
      gemm_kernel<false><<<dim3(B_ / 64, 3 * D_ / 64), 128, 0, stream>>>(
          h_bf, wt_qkv + (size_t)l * 3 * D_ * D_, qkv_b + (size_t)l * 3 * D_,
          qkv_f, nullptr, B_, 3 * D_, D_);
      scatter_kv_kernel<<<B_ * D_ / 256, 256, 0, stream>>>(
          qkv_f, kc + (size_t)l * cacheL, vc + (size_t)l * cacheL, t);
      attn_kernel<<<B_ * H_ * 32 / 256, 256, 0, stream>>>(
          qkv_f, kc + (size_t)l * cacheL, vc + (size_t)l * cacheL, ao_bf, t);
      // o @ out_w + out_b          : [512,512]x[512,512]
      gemm_kernel<false><<<dim3(B_ / 64, D_ / 64), 128, 0, stream>>>(
          ao_bf, wt_out + (size_t)l * D_ * D_, out_b + (size_t)l * D_,
          dlt, nullptr, B_, D_, D_);
      add_ln_kernel<<<B_, 128, 0, stream>>>(h_f32, dlt, ln1_s + (size_t)l * D_,
                                            ln1_b + (size_t)l * D_, h_f32, h_bf);
      // relu(h @ w1 + b1)          : [512,512]x[512,2048]
      gemm_kernel<true><<<dim3(B_ / 64, FF_ / 64), 128, 0, stream>>>(
          h_bf, wt_ff1 + (size_t)l * FF_ * D_, ff_b1 + (size_t)l * FF_,
          nullptr, ffm_bf, B_, FF_, D_);
      // mid @ w2 + b2              : [512,2048]x[2048,512]
      gemm_kernel<false><<<dim3(B_ / 64, D_ / 64), 128, 0, stream>>>(
          ffm_bf, wt_ff2 + (size_t)l * D_ * FF_, ff_b2 + (size_t)l * D_,
          dlt, nullptr, B_, D_, FF_);
      add_ln_kernel<<<B_, 128, 0, stream>>>(h_f32, dlt, ln2_s + (size_t)l * D_,
                                            ln2_b + (size_t)l * D_, h_f32, h_bf);
    }
    head_kernel<<<(B_ * 3 + 255) / 256, 256, 0, stream>>>(h_f32, head_w, head_b, out, t);
  }
}
